// ConditionalRandomField_56667798503646
// MI455X (gfx1250) — matrix operand order repaired and checked by reading, not compiled, not run on hardware
//
#include <hip/hip_runtime.h>
#include <hip/hip_bf16.h>
#include <stdint.h>

#define NUM_TAGS 512
#define START_TAG 510
#define STOP_TAG 511
#define BATCH 128
#define SEQ_LEN 512
#define NEG_INF -10000.0f

typedef __attribute__((ext_vector_type(16))) __bf16 v16bf;
typedef __attribute__((ext_vector_type(8)))  float  v8f;

static __device__ __forceinline__ unsigned short f32_to_bf16_bits(float x) {
    union { float f; uint32_t u; } cv; cv.f = x;
    uint32_t u = cv.u;
    u += 0x7FFFu + ((u >> 16) & 1u);   // round-to-nearest-even
    return (unsigned short)(u >> 16);
}

// ---------------------------------------------------------------------------
// Prep: c[i] = max_j Tr[i][j];  Wt[j][i] = bf16(exp(Tr[i][j] - c[i]))
// One block per "next" row i (512 blocks x 256 threads).
// ---------------------------------------------------------------------------
__global__ void crf_prep(const float* __restrict__ tr,
                         unsigned short* __restrict__ Wt,
                         float* __restrict__ c) {
    __shared__ float red[256];
    const int i = blockIdx.x;
    const int t = threadIdx.x;
    const float* row = tr + (size_t)i * NUM_TAGS;
    const float v0 = row[t];
    const float v1 = row[t + 256];
    red[t] = fmaxf(v0, v1);
    __syncthreads();
    for (int off = 128; off > 0; off >>= 1) {
        if (t < off) red[t] = fmaxf(red[t], red[t + off]);
        __syncthreads();
    }
    const float ci = red[0];
    if (t == 0) c[i] = ci;
    Wt[(size_t)t         * NUM_TAGS + i] = f32_to_bf16_bits(__expf(v0 - ci));
    Wt[(size_t)(t + 256) * NUM_TAGS + i] = f32_to_bf16_bits(__expf(v1 - ci));
}

__global__ void crf_zero(float* out) {
    if (threadIdx.x == 0 && blockIdx.x == 0) out[0] = 0.0f;
}

// ---------------------------------------------------------------------------
// Main persistent kernel: one block per 16-batch tile (8 blocks), 32 waves.
// Per step s: rescale alpha -> bf16 probs in LDS, then each wave computes a
// 16x16 D-tile of S = p x Wt via v_wmma_f32_16x16x32_bf16 over K=512, and
// writes new_alpha = emit + m_b + c_i + log(S). Emission loads are issued at
// the top of the step (non-temporal) so their HBM latency hides under the
// reductions and the K-loop.
// ---------------------------------------------------------------------------
__global__ void __launch_bounds__(1024)
crf_forward(const float* __restrict__ inputs,            // [B,S,T]
            const unsigned short* __restrict__ Wt,       // [T,T] bf16: Wt[prev][next]
            const float* __restrict__ c,                 // [T]
            const float* __restrict__ tr,                // [T,T] fp32 (STOP row)
            float* __restrict__ out) {
    __shared__ __align__(16) float          alpha[16][NUM_TAGS];   // 32 KB
    __shared__ __align__(16) unsigned short pb[16][NUM_TAGS];      // 16 KB
    __shared__ float mrow[16];

    const int tid  = threadIdx.x;
    const int wave = tid >> 5;
    const int lane = tid & 31;
    const int b0   = blockIdx.x * 16;

    // init alpha0
    for (int f = tid; f < 16 * NUM_TAGS; f += 1024) {
        const int b = f >> 9, j = f & 511;
        alpha[b][j] = (j == START_TAG) ? 0.0f : NEG_INF;
    }
    __syncthreads();

    const int n0     = wave * 16;                 // this wave's N tile
    const int nloc   = lane & 15;
    const int n      = n0 + nloc;                 // global next-tag column
    const float cn   = c[n];
    const int mA     = lane & 15;                 // A fragment: M row
    const int kbaseA = (lane < 16) ? 0 : 8;       // A fragment: K sub-base
    const int mbase  = (lane < 16) ? 0 : 8;       // D fragment: M sub-base

    // per-lane emission base pointer: inputs[b0 + mbase][0][n]
    const float* emitPtr =
        inputs + ((size_t)(b0 + mbase) * SEQ_LEN) * NUM_TAGS + n;

    for (int s = 0; s < SEQ_LEN; ++s) {
        // --- 0) issue this step's emission loads early (non-temporal,
        //        stream-once data; overlap with phases 1-3) ---
        float emitv[8];
        {
            const float* ep = emitPtr + (size_t)s * NUM_TAGS;
            #pragma unroll
            for (int r = 0; r < 8; ++r)
                emitv[r] = __builtin_nontemporal_load(
                    ep + (size_t)r * SEQ_LEN * NUM_TAGS);
        }

        // --- 1) per-batch max over alpha (waves 0..15, one batch each) ---
        if (wave < 16) {
            float m = -3.4e38f;
            #pragma unroll
            for (int t = 0; t < 16; ++t) m = fmaxf(m, alpha[wave][lane + 32 * t]);
            #pragma unroll
            for (int off = 16; off > 0; off >>= 1) m = fmaxf(m, __shfl_xor(m, off, 32));
            if (lane == 0) mrow[wave] = m;
        }
        __syncthreads();

        // --- 2) pb = bf16(exp(alpha - m_b)) : 8 contiguous elems / thread ---
        {
            const int f0 = tid * 8;
            const int b  = f0 >> 9, j = f0 & 511;
            const float mb = mrow[b];
            #pragma unroll
            for (int e = 0; e < 8; ++e)
                pb[b][j + e] = f32_to_bf16_bits(__expf(alpha[b][j + e] - mb));
        }
        __syncthreads();

        // --- 3) WMMA: D[m][n] = sum_k pb[m][k] * Wt[k][n] ---
        v8f acc = {};
        #pragma unroll 4
        for (int k0 = 0; k0 < NUM_TAGS; k0 += 32) {
            union { uint4 q[2]; v16bf v; } A, B;
            // A 16x32 bf16: lane<16 holds K {k0..k0+7, k0+16..k0+23}; lane>=16 +8
            A.q[0] = *(const uint4*)&pb[mA][k0 + kbaseA];
            A.q[1] = *(const uint4*)&pb[mA][k0 + 16 + kbaseA];
            // B 32x16 bf16: lane holds column N = n, K {k0..k0+7, k0+16..k0+23}; lane>=16 +8
            const unsigned short* bcol = Wt + (size_t)(k0 + kbaseA) * NUM_TAGS + n;
            unsigned short* bs = (unsigned short*)B.q;
            #pragma unroll
            for (int e = 0; e < 8; ++e) {
                bs[e]     = bcol[(size_t)e * NUM_TAGS];
                bs[8 + e] = bcol[(size_t)(16 + e) * NUM_TAGS];
            }
            acc = __builtin_amdgcn_wmma_f32_16x16x32_bf16(
                false, A.v, false, B.v, (short)0, acc, false, false);
        }

        // --- write back: D element r -> (m = r + mbase, n) ---
        #pragma unroll
        for (int r = 0; r < 8; ++r) {
            const int m = r + mbase;
            alpha[m][n] = emitv[r] + mrow[m] + cn + __logf(fmaxf(acc[r], 1e-37f));
        }
        __syncthreads();
    }

    // --- terminal: loss_b = logsumexp_i(alpha[b][i] + Tr[STOP][i]) ---
    if (wave < 16) {
        const float* trStop = tr + (size_t)STOP_TAG * NUM_TAGS;
        float vals[16];
        float m = -3.4e38f;
        #pragma unroll
        for (int t = 0; t < 16; ++t) {
            const int j = lane + 32 * t;
            vals[t] = alpha[wave][j] + trStop[j];
            m = fmaxf(m, vals[t]);
        }
        #pragma unroll
        for (int off = 16; off > 0; off >>= 1) m = fmaxf(m, __shfl_xor(m, off, 32));
        float ssum = 0.0f;
        #pragma unroll
        for (int t = 0; t < 16; ++t) ssum += __expf(vals[t] - m);
        #pragma unroll
        for (int off = 16; off > 0; off >>= 1) ssum += __shfl_xor(ssum, off, 32);
        if (lane == 0) atomicAdd(out, m + __logf(ssum));
    }
}

// ---------------------------------------------------------------------------
extern "C" void kernel_launch(void* const* d_in, const int* in_sizes, int n_in,
                              void* d_out, int out_size, void* d_ws, size_t ws_size,
                              hipStream_t stream) {
    (void)in_sizes; (void)n_in; (void)out_size; (void)ws_size;
    const float* inputs      = (const float*)d_in[0];
    // d_in[1] = tags (unused by the reference forward)
    const float* transitions = (const float*)d_in[2];

    unsigned short* Wt = (unsigned short*)d_ws;                       // 512*512*2 = 512 KB
    float* c = (float*)((char*)d_ws +
                        (size_t)NUM_TAGS * NUM_TAGS * sizeof(unsigned short)); // +2 KB
    float* out = (float*)d_out;

    crf_prep<<<NUM_TAGS, 256, 0, stream>>>(transitions, Wt, c);
    crf_zero<<<1, 32, 0, stream>>>(out);
    crf_forward<<<BATCH / 16, 1024, 0, stream>>>(inputs, Wt, c, transitions, out);
}
